// MLP_mia_white2_65300682768664
// MI455X (gfx1250) — compile-verified, matrix operand order checked by reading, not run on hardware
//
#include <hip/hip_runtime.h>

typedef __attribute__((ext_vector_type(16))) _Float16 v16h;
typedef __attribute__((ext_vector_type(8)))  float    v8f;

#define NL      3        // layers
#define NNODES  200000
#define DIM     128
#define H0N     128
#define H1N     64
#define EPSF    1e-8f

#define WAVES_PER_BLOCK 8
#define EDGES_PER_WAVE  16
#define EDGES_PER_BLOCK (WAVES_PER_BLOCK * EDGES_PER_WAVE)

__global__ __launch_bounds__(256)
void fused_edge_mlp(const int* __restrict__ edge_list,
                    const float* __restrict__ z1,
                    const float* __restrict__ z2,
                    const float* __restrict__ w1,
                    const float* __restrict__ w2,
                    const float* __restrict__ W0,
                    const float* __restrict__ b0,
                    const float* __restrict__ W1,
                    const float* __restrict__ b1,
                    const float* __restrict__ Wp,
                    const float* __restrict__ bp,
                    float* __restrict__ out,
                    int nE)
{
    // per-wave private staging: h (16x128 f16) and h2 (16x64 f16)
    __shared__ _Float16 sh_h [WAVES_PER_BLOCK][EDGES_PER_WAVE * H0N]; // 32 KB
    __shared__ _Float16 sh_h2[WAVES_PER_BLOCK][EDGES_PER_WAVE * H1N]; // 16 KB

    const int tid  = threadIdx.x;
    const int wv   = tid >> 5;
    const int lane = tid & 31;
    const int half = lane >> 4;   // which 64-dim half of D this lane reduces
    const int m    = lane & 15;   // edge-in-wave / matrix row / matrix col

    const int edge = blockIdx.x * EDGES_PER_BLOCK + wv * EDGES_PER_WAVE + m;
    const int ce   = edge < nE ? edge : nE - 1;

    const int e0 = edge_list[2 * ce + 0];
    const int e1 = edge_list[2 * ce + 1];

    // ---------------- Phase 1: edge features (cos, dot) x 3 layers x 2 trains
    float sim[6];
#pragma unroll
    for (int i = 0; i < 6; ++i) sim[i] = 0.f;

#pragma unroll
    for (int t = 0; t < 2; ++t) {
        const float* zt = (t == 0) ? z1 : z2;
        const float* wt = (t == 0) ? w1 : w2;
        const float4* wp4 = (const float4*)wt;
#pragma unroll
        for (int l = 0; l < NL; ++l) {
            const float4* pa = (const float4*)(zt + ((size_t)l * NNODES + (size_t)e0) * DIM);
            const float4* pb = (const float4*)(zt + ((size_t)l * NNODES + (size_t)e1) * DIM);
            float pd = 0.f, qa = 0.f, qb = 0.f;
#pragma unroll 4
            for (int i = 0; i < 16; ++i) {
                const int idx = (half << 4) + i;   // this lane's 64-dim half
                const float4 wv4 = wp4[idx];
                const float4 av  = pa[idx];
                const float4 bv  = pb[idx];
                const float ax = wv4.x * av.x, ay = wv4.y * av.y,
                            az = wv4.z * av.z, aw = wv4.w * av.w;
                const float bx = wv4.x * bv.x, by = wv4.y * bv.y,
                            bz = wv4.z * bv.z, bw = wv4.w * bv.w;
                pd += ax * bx + ay * by + az * bz + aw * bw;
                qa += ax * ax + ay * ay + az * az + aw * aw;
                qb += bx * bx + by * by + bz * bz + bw * bw;
            }
            // combine the two D-halves (lanes L and L^16 hold the same edge)
            pd += __shfl_xor(pd, 16, 32);
            qa += __shfl_xor(qa, 16, 32);
            qb += __shfl_xor(qb, 16, 32);
            const float n0 = fmaxf(sqrtf(qa), EPSF);
            const float n1 = fmaxf(sqrtf(qb), EPSF);
            sim[2 * l]     += pd / (n0 * n1);  // cos
            sim[2 * l + 1] += pd;              // dot
        }
    }

    // ---------------- Phase 2: layer0  h = relu(sim @ W0^T + b0), WMMA f16
    // A frag (16x32, K padded): lanes 0-15 carry k=0..7,16..23 -> only k<6 live
    v16h afrag = {};
    if (half == 0) {
        afrag[0] = (_Float16)sim[0];
        afrag[1] = (_Float16)sim[1];
        afrag[2] = (_Float16)sim[2];
        afrag[3] = (_Float16)sim[3];
        afrag[4] = (_Float16)sim[4];
        afrag[5] = (_Float16)sim[5];
    }
#pragma unroll
    for (int nt = 0; nt < H0N / 16; ++nt) {
        v16h bfrag = {};
        if (half == 0) {
            const float* wrow = &W0[(nt * 16 + m) * 6];   // B[k][n] = W0[n][k]
#pragma unroll
            for (int i = 0; i < 6; ++i) bfrag[i] = (_Float16)wrow[i];
        }
        v8f c = {};
        c = __builtin_amdgcn_wmma_f32_16x16x32_f16(false, afrag, false, bfrag,
                                                   (short)0, c, false, false);
        const float bias = b0[nt * 16 + m];
#pragma unroll
        for (int r = 0; r < 8; ++r) {
            const int row = r + 8 * half;
            float v = c[r] + bias;
            v = v > 0.f ? v : 0.f;
            sh_h[wv][row * H0N + nt * 16 + m] = (_Float16)v;
        }
    }
    __syncthreads();

    // ---------------- Phase 3: layer1  h2 = relu(h @ W1^T + b1), 16 WMMAs
    v8f acc[4];
#pragma unroll
    for (int nt = 0; nt < 4; ++nt) { v8f zz = {}; acc[nt] = zz; }

#pragma unroll
    for (int kc = 0; kc < 4; ++kc) {
        v16h a;
        const _Float16* hrow = &sh_h[wv][m * H0N];
#pragma unroll
        for (int i = 0; i < 16; ++i) {
            const int k = kc * 32 + (i & 7) + (half << 3) + ((i >> 3) << 4);
            a[i] = hrow[k];
        }
#pragma unroll
        for (int nt = 0; nt < 4; ++nt) {
            v16h b;
            const float* w1row = &W1[(nt * 16 + m) * H0N];   // B[k][n] = W1[n][k]
#pragma unroll
            for (int i = 0; i < 16; ++i) {
                const int k = kc * 32 + (i & 7) + (half << 3) + ((i >> 3) << 4);
                b[i] = (_Float16)w1row[k];
            }
            acc[nt] = __builtin_amdgcn_wmma_f32_16x16x32_f16(false, a, false, b,
                                                             (short)0, acc[nt],
                                                             false, false);
        }
    }

#pragma unroll
    for (int nt = 0; nt < 4; ++nt) {
        const float bias = b1[nt * 16 + m];
#pragma unroll
        for (int r = 0; r < 8; ++r) {
            const int row = r + 8 * half;
            float v = acc[nt][r] + bias;
            v = v > 0.f ? v : 0.f;
            sh_h2[wv][row * H1N + nt * 16 + m] = (_Float16)v;
        }
    }
    __syncthreads();

    // ---------------- Phase 4: out = h2 @ Wp^T + bp   (per-lane f32 dot)
    if (half == 0) {
        float a = bp[0];
        const _Float16* h2row = &sh_h2[wv][m * H1N];
#pragma unroll
        for (int j = 0; j < H1N; ++j) a += Wp[j] * (float)h2row[j];
        if (edge < nE) out[edge] = a;
    }
}

extern "C" void kernel_launch(void* const* d_in, const int* in_sizes, int n_in,
                              void* d_out, int out_size, void* d_ws, size_t ws_size,
                              hipStream_t stream)
{
    const int*   edge_list = (const int*)  d_in[0];
    const float* z1        = (const float*)d_in[1];
    const float* z2        = (const float*)d_in[2];
    const float* w1        = (const float*)d_in[3];
    const float* w2        = (const float*)d_in[4];
    const float* W0        = (const float*)d_in[5];
    const float* b0        = (const float*)d_in[6];
    const float* W1        = (const float*)d_in[7];
    const float* b1        = (const float*)d_in[8];
    const float* Wp        = (const float*)d_in[9];
    const float* bp        = (const float*)d_in[10];
    float*       out       = (float*)d_out;

    const int nE = in_sizes[0] / 2;
    const int blocks = (nE + EDGES_PER_BLOCK - 1) / EDGES_PER_BLOCK;

    fused_edge_mlp<<<blocks, 256, 0, stream>>>(edge_list, z1, z2, w1, w2,
                                               W0, b0, W1, b1, Wp, bp, out, nE);
}
